// H3_PI_88098369176186
// MI455X (gfx1250) — matrix operand order repaired and checked by reading, not compiled, not run on hardware
//
#include <hip/hip_runtime.h>
#include <cstdint>

// ---------------- problem constants (match reference) ----------------
#define NA     50000
#define NC2    10000
#define NC3    2000
#define EBT    800000
#define EI2    160000
#define EI3    32000
#define IN_DIM 128
#define HID    256
#define HEADS  4
#define DH     64
#define PI_DIM 64
#define BN_EPS 1e-5f
#define NEG_SLOPE 0.2f

typedef __bf16 v16bf __attribute__((ext_vector_type(16)));
typedef __bf16 v8bf  __attribute__((ext_vector_type(8)));
typedef __bf16 v2bf  __attribute__((ext_vector_type(2)));
typedef float  v8f   __attribute__((ext_vector_type(8)));

static inline int nblk(long long n, int bs) {
    long long b = (n + bs - 1) / bs;
    if (b > 2147483000LL) b = 2147483000LL;
    return (int)b;
}

// ---------------- ordered-float <-> uint mapping for atomic max ----------------
__device__ __forceinline__ unsigned f2o(float f) {
    unsigned u = __float_as_uint(f);
    return (u & 0x80000000u) ? ~u : (u | 0x80000000u);
}
__device__ __forceinline__ float o2f(unsigned u) {
    if (u == 0u) return 0.0f;  // empty-segment sentinel -> reference's where(isfinite, m, 0)
    return __uint_as_float((u & 0x80000000u) ? (u ^ 0x80000000u) : ~u);
}

// =====================================================================
// WMMA bf16 GEMM:  C[m, coff+n] = (relu?)(A(MxK) @ W(KxN) + bias)
// block = 128 threads (4 waves), tile 64(M) x 64(N), K step 32.
// LDS tiles are stored in *fragment-major* order so each lane's 16-elem
// bf16 fragment is 32 contiguous bytes -> 2x ds_load_b128 per fragment.
//   A frag (16x32): element e <-> k = hh*16 + e,                lane = hh*16 + (m&15)
//   B frag (32x16): element e <-> k = hh*16 + e,                lane = hh*16 + (n&15)
// K must be a multiple of 32, N a multiple of 64 (true for all calls).
// =====================================================================
#define FRAG_ST 24   // per-lane fragment stride in bf16 elems (16 used + 8 pad, keeps 16B align)

__global__ __launch_bounds__(128)
void wmma_gemm_kernel(const float* __restrict__ A, int M, int K,
                      const float* __restrict__ W, int N,
                      const float* __restrict__ bias,
                      float* __restrict__ C, int ldc, int coff, int relu)
{
    __shared__ __bf16 AsP[4 * 32 * FRAG_ST] __attribute__((aligned(32)));  // 4 wave-slots
    __shared__ __bf16 BsP[4 * 32 * FRAG_ST] __attribute__((aligned(32)));  // 4 n-subtiles

    const int tid  = threadIdx.x;
    const int wave = tid >> 5;
    const int lane = tid & 31;
    const int m0   = blockIdx.x * 64;
    const int n0   = blockIdx.y * 64;

    v8f acc[4];
#pragma unroll
    for (int nt = 0; nt < 4; ++nt) acc[nt] = (v8f){0.f,0.f,0.f,0.f,0.f,0.f,0.f,0.f};

    for (int k0 = 0; k0 < K; k0 += 32) {
        // ---- stage A tile (64 rows x 32 k) in fragment order ----
        // 256 groups of 8 consecutive k; each group is one contiguous e-run of a lane slot.
#pragma unroll
        for (int gi = 0; gi < 2; ++gi) {
            int g    = tid + (gi << 7);      // 0..255
            int row  = g >> 2;               // 0..63
            int kg   = g & 3;                // 8-k group
            int w    = row >> 4, lrow = row & 15;
            int hh   = kg >> 1;
            int eb   = (kg & 1) << 3;        // element base 0 or 8
            __bf16* dst = AsP + ((w << 5) + (hh << 4) + lrow) * FRAG_ST + eb;
            int gm = m0 + row;
            v8bf pk;
            if (gm < M) {
                const float* ap = A + (size_t)gm * K + k0 + (kg << 3);
                if (k0 + 32 < K) __builtin_prefetch(ap + 32, 0, 3);  // global_prefetch_b8
#pragma unroll
                for (int j = 0; j < 8; ++j) pk[j] = (__bf16)ap[j];
            } else {
#pragma unroll
                for (int j = 0; j < 8; ++j) pk[j] = (__bf16)0.0f;
            }
            *(v8bf*)dst = pk;                // ds_store_b128
        }
        // ---- stage B tile (32 k x 64 n) in fragment order ----
        // thread: k-pair (kp) x 8-wide n chunk; packs (k, k+1) into one b32 store.
        {
            int kp  = tid >> 3;              // 0..15
            int nc  = tid & 7;               // 0..7
            int k   = kp << 1;
            int n0l = nc << 3;
            const float* w0 = W + (size_t)(k0 + k) * N + n0 + n0l;
            const float* w1 = w0 + N;
#pragma unroll
            for (int j = 0; j < 8; ++j) {
                int n      = n0l + j;
                int nt     = n >> 4;
                int lane_s = ((k >> 4) << 4) + (n & 15);
                v2bf p;
                p[0] = (__bf16)w0[j];
                p[1] = (__bf16)w1[j];
                *(v2bf*)(BsP + ((nt << 5) + lane_s) * FRAG_ST + (k & 15)) = p;  // ds_store_b32
            }
        }
        __syncthreads();

        v16bf afrag = *(const v16bf*)(AsP + ((wave << 5) + lane) * FRAG_ST);  // 2x ds_load_b128
#pragma unroll
        for (int nt = 0; nt < 4; ++nt) {
            v16bf bfrag = *(const v16bf*)(BsP + ((nt << 5) + lane) * FRAG_ST);
            acc[nt] = __builtin_amdgcn_wmma_f32_16x16x32_bf16(
                false, afrag, false, bfrag, (short)0, acc[nt], false, false);
        }
        __syncthreads();
    }

    // ---- store: C/D layout m = i + 8*hh, n = lane&15 ----
    const int hh = lane >> 4;
    const int nn = lane & 15;
#pragma unroll
    for (int nt = 0; nt < 4; ++nt) {
        int col = n0 + (nt << 4) + nn;
        float bv = bias ? bias[col] : 0.0f;
#pragma unroll
        for (int i = 0; i < 8; ++i) {
            int m = m0 + (wave << 4) + (hh << 3) + i;
            if (m < M) {
                float v = acc[nt][i] + bv;
                if (relu) v = fmaxf(v, 0.0f);
                C[(size_t)m * ldc + coff + col] = v;
            }
        }
    }
}

// =====================================================================
// elementwise / scatter kernels
// =====================================================================
__global__ void zero_f(float* p, long long n) {
    for (long long i = (long long)blockIdx.x * blockDim.x + threadIdx.x; i < n;
         i += (long long)gridDim.x * blockDim.x) p[i] = 0.0f;
}
__global__ void zero_u(unsigned* p, long long n) {
    for (long long i = (long long)blockIdx.x * blockDim.x + threadIdx.x; i < n;
         i += (long long)gridDim.x * blockDim.x) p[i] = 0u;
}
__global__ void copy_f(const float* __restrict__ s, float* __restrict__ d, long long n) {
    for (long long i = (long long)blockIdx.x * blockDim.x + threadIdx.x; i < n;
         i += (long long)gridDim.x * blockDim.x) d[i] = s[i];
}

// el/er per (node, head)
__global__ void gat_attn_coef(const float* __restrict__ feat,
                              const float* __restrict__ al, const float* __restrict__ ar,
                              float* __restrict__ el, float* __restrict__ er, int n)
{
    int idx = blockIdx.x * blockDim.x + threadIdx.x;
    if (idx >= n * HEADS) return;
    int i = idx >> 2, h = idx & 3;
    const float* f = feat + (size_t)i * HID + h * DH;
    float sl = 0.f, sr = 0.f;
    for (int d = 0; d < DH; ++d) { float v = f[d]; sl += v * al[h * DH + d]; sr += v * ar[h * DH + d]; }
    el[idx] = sl; er[idx] = sr;
}

__device__ __forceinline__ float lrelu(float x) { return x > 0.f ? x : NEG_SLOPE * x; }

__global__ void gat_edge_max(const float* __restrict__ el, const float* __restrict__ er,
                             const int* __restrict__ src, const int* __restrict__ dst,
                             unsigned* __restrict__ mu, int E)
{
    long long n = (long long)E * HEADS;
    for (long long idx = (long long)blockIdx.x * blockDim.x + threadIdx.x; idx < n;
         idx += (long long)gridDim.x * blockDim.x) {
        int e = (int)(idx >> 2), h = (int)(idx & 3);
        int s = src[e], t = dst[e];
        float x = lrelu(el[s * 4 + h] + er[t * 4 + h]);
        atomicMax(&mu[t * 4 + h], f2o(x));
    }
}
__global__ void decode_max(const unsigned* __restrict__ mu, float* __restrict__ m, long long n) {
    for (long long i = (long long)blockIdx.x * blockDim.x + threadIdx.x; i < n;
         i += (long long)gridDim.x * blockDim.x) m[i] = o2f(mu[i]);
}
__global__ void gat_edge_den(const float* __restrict__ el, const float* __restrict__ er,
                             const float* __restrict__ m,
                             const int* __restrict__ src, const int* __restrict__ dst,
                             float* __restrict__ den, int E)
{
    long long n = (long long)E * HEADS;
    for (long long idx = (long long)blockIdx.x * blockDim.x + threadIdx.x; idx < n;
         idx += (long long)gridDim.x * blockDim.x) {
        int e = (int)(idx >> 2), h = (int)(idx & 3);
        int s = src[e], t = dst[e];
        float x = lrelu(el[s * 4 + h] + er[t * 4 + h]);
        atomicAdd(&den[t * 4 + h], __expf(x - m[t * 4 + h]));
    }
}
// out[dst, h*64+d] += alpha * feat[src, h*64+d]; one thread per (edge, d), loops heads
__global__ void gat_edge_agg(const float* __restrict__ feat,
                             const float* __restrict__ el, const float* __restrict__ er,
                             const float* __restrict__ m, const float* __restrict__ den,
                             const int* __restrict__ src, const int* __restrict__ dst,
                             float* __restrict__ out, int E)
{
    long long n = (long long)E * DH;
    for (long long idx = (long long)blockIdx.x * blockDim.x + threadIdx.x; idx < n;
         idx += (long long)gridDim.x * blockDim.x) {
        int e = (int)(idx >> 6), d = (int)(idx & 63);
        int s = src[e], t = dst[e];
#pragma unroll
        for (int h = 0; h < HEADS; ++h) {
            float x = lrelu(el[s * 4 + h] + er[t * 4 + h]);
            float ex = __expf(x - m[t * 4 + h]);
            float alpha = ex / fmaxf(den[t * 4 + h], 1e-9f);
            atomicAdd(&out[(size_t)t * HID + h * DH + d],
                      alpha * feat[(size_t)s * HID + h * DH + d]);
        }
    }
}

// xcat[i] = [relu(gat_out[i]+b) | feats[i]]
__global__ void build_xcat(const float* __restrict__ gat_out, const float* __restrict__ b,
                           const float* __restrict__ feats, float* __restrict__ xcat, int n)
{
    long long tot = (long long)n * (HID + IN_DIM);
    for (long long idx = (long long)blockIdx.x * blockDim.x + threadIdx.x; idx < tot;
         idx += (long long)gridDim.x * blockDim.x) {
        int i = (int)(idx / (HID + IN_DIM)), d = (int)(idx % (HID + IN_DIM));
        float v = (d < HID) ? fmaxf(gat_out[(size_t)i * HID + d] + b[d], 0.0f)
                            : feats[(size_t)i * IN_DIM + (d - HID)];
        xcat[idx] = v;
    }
}

// out[dst[e]*D + d] += X[src[e]*D + d]  (segment_sum)
__global__ void seg_sum_edges(const float* __restrict__ X,
                              const int* __restrict__ src, const int* __restrict__ dst,
                              float* __restrict__ out, int E, int D)
{
    long long n = (long long)E * D;
    for (long long idx = (long long)blockIdx.x * blockDim.x + threadIdx.x; idx < n;
         idx += (long long)gridDim.x * blockDim.x) {
        int e = (int)(idx / D), d = (int)(idx % D);
        atomicAdd(&out[(size_t)dst[e] * D + d], X[(size_t)src[e] * D + d]);
    }
}

// batch-norm stats per column (D = HID = 256), biased variance
__global__ __launch_bounds__(256)
void bn_stats(const float* __restrict__ X, int M, float* __restrict__ stats)
{
    int col = blockIdx.x;
    float s = 0.f, s2 = 0.f;
    for (int r = threadIdx.x; r < M; r += 256) {
        float v = X[(size_t)r * HID + col];
        s += v; s2 += v * v;
    }
    __shared__ float sh[256], sh2[256];
    sh[threadIdx.x] = s; sh2[threadIdx.x] = s2;
    __syncthreads();
    for (int o = 128; o > 0; o >>= 1) {
        if (threadIdx.x < o) { sh[threadIdx.x] += sh[threadIdx.x + o]; sh2[threadIdx.x] += sh2[threadIdx.x + o]; }
        __syncthreads();
    }
    if (threadIdx.x == 0) {
        float mu = sh[0] / (float)M;
        stats[col] = mu;
        stats[HID + col] = sh2[0] / (float)M - mu * mu;
    }
}
__global__ void bn_apply(float* __restrict__ X, int M, const float* __restrict__ stats,
                         const float* __restrict__ g, const float* __restrict__ beta)
{
    long long n = (long long)M * HID;
    for (long long idx = (long long)blockIdx.x * blockDim.x + threadIdx.x; idx < n;
         idx += (long long)gridDim.x * blockDim.x) {
        int c = (int)(idx % HID);
        float v = (X[idx] - stats[c]) * rsqrtf(stats[HID + c] + BN_EPS) * g[c] + beta[c];
        X[idx] = fmaxf(v, 0.0f);
    }
}

// segment_max pool: mu[dst[i]*D+d] = max(..., X[src[i]*ldx + d])
__global__ void pool_max(const float* __restrict__ X, int ldx,
                         const int* __restrict__ src, const int* __restrict__ dst,
                         unsigned* __restrict__ mu, int Nsrc, int D)
{
    long long n = (long long)Nsrc * D;
    for (long long idx = (long long)blockIdx.x * blockDim.x + threadIdx.x; idx < n;
         idx += (long long)gridDim.x * blockDim.x) {
        int i = (int)(idx / D), d = (int)(idx % D);
        atomicMax(&mu[(size_t)dst[i] * D + d], f2o(X[(size_t)src[i] * ldx + d]));
    }
}
// out[i] = [decode(mu[i]) | pimg[i]], row width D+DP
__global__ void pool_decode_concat(const unsigned* __restrict__ mu, const float* __restrict__ pimg,
                                   float* __restrict__ out, int N, int D, int DP)
{
    long long n = (long long)N * (D + DP);
    for (long long idx = (long long)blockIdx.x * blockDim.x + threadIdx.x; idx < n;
         idx += (long long)gridDim.x * blockDim.x) {
        int i = (int)(idx / (D + DP)), d = (int)(idx % (D + DP));
        out[idx] = (d < D) ? o2f(mu[(size_t)i * D + d]) : pimg[(size_t)i * DP + (d - D)];
    }
}

// per-column sum readout
__global__ __launch_bounds__(256)
void colsum(const float* __restrict__ X, int M, int D, float* __restrict__ out)
{
    int col = blockIdx.x;
    float s = 0.f;
    for (int r = threadIdx.x; r < M; r += 256) s += X[(size_t)r * D + col];
    __shared__ float sh[256];
    sh[threadIdx.x] = s;
    __syncthreads();
    for (int o = 128; o > 0; o >>= 1) {
        if (threadIdx.x < o) sh[threadIdx.x] += sh[threadIdx.x + o];
        __syncthreads();
    }
    if (threadIdx.x == 0) out[col] = sh[0];
}

// final MLP: relu(r(1x1024) @ W1(1024x256) + b1) @ W2(256x1) + b2
__global__ __launch_bounds__(256)
void out_mlp(const float* __restrict__ r, const float* __restrict__ W1, const float* __restrict__ b1,
             const float* __restrict__ W2, const float* __restrict__ b2, float* __restrict__ out)
{
    int j = threadIdx.x;
    float s = b1[j];
    for (int k = 0; k < 4 * HID; ++k) s += r[k] * W1[k * HID + j];
    s = fmaxf(s, 0.0f);
    __shared__ float sh[256];
    sh[j] = s * W2[j];
    __syncthreads();
    for (int o = 128; o > 0; o >>= 1) {
        if (j < o) sh[j] += sh[j + o];
        __syncthreads();
    }
    if (j == 0) out[0] = sh[0] + b2[0];
}

// =====================================================================
// host orchestration
// =====================================================================
static void launch_gemm(const float* A, int M, int K, const float* W, int N,
                        const float* bias, float* C, int ldc, int coff, int relu,
                        hipStream_t stream)
{
    dim3 grid((M + 63) / 64, N / 64);
    wmma_gemm_kernel<<<grid, 128, 0, stream>>>(A, M, K, W, N, bias, C, ldc, coff, relu);
}

extern "C" void kernel_launch(void* const* d_in, const int* in_sizes, int n_in,
                              void* d_out, int out_size, void* d_ws, size_t ws_size,
                              hipStream_t stream)
{
    (void)in_sizes; (void)n_in; (void)out_size; (void)ws_size;
    // Input order: setup_inputs() insertion order; "params" flattened in JAX pytree
    // (alphabetical dict key) leaf order:
    //  0 feats, 1 pimg2, 2 pimg3,
    //  3..12  bt[0]: gat.W, gat.al, gat.ar, gat.b, gin.beta, gin.g, gin.l1.W, gin.l1.b, gin.l2.W, gin.l2.b
    //  13..22 bt[1]: same
    //  23..28 h2[0]: beta, g, l1.W, l1.b, l2.W, l2.b ; 29..34 h2[1]
    //  35..40 h3[0] ; 41..46 h3[1]
    //  47 out.l1.W, 48 out.l1.b, 49 out.l2.W, 50 out.l2.b,
    //  51 e1_src, 52 e1_dst, 53 e2_src, 54 e2_dst,
    //  55 g1_src, 56 g1_dst, 57 i2_src, 58 i2_dst, 59 g2_src, 60 g2_dst, 61 i3_src, 62 i3_dst
    const float* feats = (const float*)d_in[0];
    const float* pimg2 = (const float*)d_in[1];
    const float* pimg3 = (const float*)d_in[2];
    auto F = [&](int i) { return (const float*)d_in[i]; };
    auto I = [&](int i) { return (const int*)d_in[i]; };

    // ---- workspace layout ----
    float* ws = (float*)d_ws;
    size_t o = 0;
    float* bufA = ws + o; o += (size_t)NA * 384;    // xcat / pooled concat
    float* bufB = ws + o; o += (size_t)NA * 384;    // x + agg
    float* bufC = ws + o; o += (size_t)NA * 256;    // GAT feat / layer outputs (aliased as pool uint buf)
    float* bufD = ws + o; o += (size_t)NA * 256;    // gat_out / hidden h1
    float* hA   = ws + o; o += (size_t)NA * 512;
    float* hC2  = ws + o; o += (size_t)NC2 * 256;
    float* hC3  = ws + o; o += (size_t)NC3 * 256;
    float* el   = ws + o; o += (size_t)NA * HEADS;
    float* er   = ws + o; o += (size_t)NA * HEADS;
    float* em   = ws + o; o += (size_t)NA * HEADS;
    float* den  = ws + o; o += (size_t)NA * HEADS;
    unsigned* emu = (unsigned*)(ws + o); o += (size_t)NA * HEADS;
    float* stats = ws + o; o += 512;
    float* red   = ws + o; o += 1024;
    unsigned* poolu = (unsigned*)bufC;  // pool phases only (bufC free at those points)

    // ================= bottom blocks (two edge types) =================
    for (int t = 0; t < 2; ++t) {
        int pb = 3 + t * 10;
        const int* esrc = I(51 + t * 2);
        const int* edst = I(52 + t * 2);

        // GAT linear: feat = feats @ W  (50000x128 @ 128x256) -> bufC
        launch_gemm(feats, NA, IN_DIM, F(pb + 0), HID, nullptr, bufC, HID, 0, 0, stream);
        // attention coefficients
        gat_attn_coef<<<nblk((long long)NA * HEADS, 256), 256, 0, stream>>>(bufC, F(pb + 1), F(pb + 2), el, er, NA);
        // edge softmax
        zero_u<<<nblk((long long)NA * HEADS, 256), 256, 0, stream>>>(emu, (long long)NA * HEADS);
        zero_f<<<nblk((long long)NA * HEADS, 256), 256, 0, stream>>>(den, (long long)NA * HEADS);
        gat_edge_max<<<nblk((long long)EBT * HEADS, 256), 256, 0, stream>>>(el, er, esrc, edst, emu, EBT);
        decode_max<<<nblk((long long)NA * HEADS, 256), 256, 0, stream>>>(emu, em, (long long)NA * HEADS);
        gat_edge_den<<<nblk((long long)EBT * HEADS, 256), 256, 0, stream>>>(el, er, em, esrc, edst, den, EBT);
        // weighted aggregation -> bufD
        zero_f<<<nblk((long long)NA * HID, 256), 256, 0, stream>>>(bufD, (long long)NA * HID);
        gat_edge_agg<<<nblk((long long)EBT * DH, 256), 256, 0, stream>>>(bufC, el, er, em, den, esrc, edst, bufD, EBT);
        // skip concat: bufA = [relu(gat_out + b) | feats]  (50000x384)
        build_xcat<<<nblk((long long)NA * 384, 256), 256, 0, stream>>>(bufD, F(pb + 3), feats, bufA, NA);
        // GIN: bufB = x + segment_sum(x[src])
        copy_f<<<nblk((long long)NA * 384, 256), 256, 0, stream>>>(bufA, bufB, (long long)NA * 384);
        seg_sum_edges<<<nblk((long long)EBT * 384, 256), 256, 0, stream>>>(bufA, esrc, edst, bufB, EBT, 384);
        // mlp_bn: l1 -> BN -> relu -> l2 (+relu from gin_conv) into hA[:, t*256:]
        launch_gemm(bufB, NA, 384, F(pb + 6), HID, F(pb + 7), bufD, HID, 0, 0, stream);
        bn_stats<<<HID, 256, 0, stream>>>(bufD, NA, stats);
        bn_apply<<<nblk((long long)NA * HID, 256), 256, 0, stream>>>(bufD, NA, stats, F(pb + 5), F(pb + 4));
        launch_gemm(bufD, NA, HID, F(pb + 8), HID, F(pb + 9), hA, 512, t * HID, 1, stream);
    }

    // ================= G1: max-pool A -> C2, concat pimg2 =================
    zero_u<<<nblk((long long)NC2 * 512, 256), 256, 0, stream>>>(poolu, (long long)NC2 * 512);
    pool_max<<<nblk((long long)NA * 512, 256), 256, 0, stream>>>(hA, 512, I(55), I(56), poolu, NA, 512);
    pool_decode_concat<<<nblk((long long)NC2 * 576, 256), 256, 0, stream>>>(poolu, pimg2, bufA, NC2, 512, PI_DIM);

    // h2[0]: GIN on 10000x576 over i2 edges
    copy_f<<<nblk((long long)NC2 * 576, 256), 256, 0, stream>>>(bufA, bufB, (long long)NC2 * 576);
    seg_sum_edges<<<nblk((long long)EI2 * 576, 256), 256, 0, stream>>>(bufA, I(57), I(58), bufB, EI2, 576);
    launch_gemm(bufB, NC2, 576, F(25), HID, F(26), bufD, HID, 0, 0, stream);
    bn_stats<<<HID, 256, 0, stream>>>(bufD, NC2, stats);
    bn_apply<<<nblk((long long)NC2 * HID, 256), 256, 0, stream>>>(bufD, NC2, stats, F(24), F(23));
    launch_gemm(bufD, NC2, HID, F(27), HID, F(28), bufC, HID, 0, 1, stream);

    // h2[1]: GIN on 10000x256 -> hC2
    copy_f<<<nblk((long long)NC2 * HID, 256), 256, 0, stream>>>(bufC, bufB, (long long)NC2 * HID);
    seg_sum_edges<<<nblk((long long)EI2 * HID, 256), 256, 0, stream>>>(bufC, I(57), I(58), bufB, EI2, HID);
    launch_gemm(bufB, NC2, HID, F(31), HID, F(32), bufD, HID, 0, 0, stream);
    bn_stats<<<HID, 256, 0, stream>>>(bufD, NC2, stats);
    bn_apply<<<nblk((long long)NC2 * HID, 256), 256, 0, stream>>>(bufD, NC2, stats, F(30), F(29));
    launch_gemm(bufD, NC2, HID, F(33), HID, F(34), hC2, HID, 0, 1, stream);

    // ================= G2: max-pool C2 -> C3, concat pimg3 =================
    zero_u<<<nblk((long long)NC3 * 256, 256), 256, 0, stream>>>(poolu, (long long)NC3 * 256);
    pool_max<<<nblk((long long)NC2 * 256, 256), 256, 0, stream>>>(hC2, 256, I(59), I(60), poolu, NC2, 256);
    pool_decode_concat<<<nblk((long long)NC3 * 320, 256), 256, 0, stream>>>(poolu, pimg3, bufA, NC3, 256, PI_DIM);

    // h3[0]: GIN on 2000x320 over i3 edges
    copy_f<<<nblk((long long)NC3 * 320, 256), 256, 0, stream>>>(bufA, bufB, (long long)NC3 * 320);
    seg_sum_edges<<<nblk((long long)EI3 * 320, 256), 256, 0, stream>>>(bufA, I(61), I(62), bufB, EI3, 320);
    launch_gemm(bufB, NC3, 320, F(37), HID, F(38), bufD, HID, 0, 0, stream);
    bn_stats<<<HID, 256, 0, stream>>>(bufD, NC3, stats);
    bn_apply<<<nblk((long long)NC3 * HID, 256), 256, 0, stream>>>(bufD, NC3, stats, F(36), F(35));
    launch_gemm(bufD, NC3, HID, F(39), HID, F(40), bufC, HID, 0, 1, stream);

    // h3[1]: GIN on 2000x256 -> hC3
    copy_f<<<nblk((long long)NC3 * HID, 256), 256, 0, stream>>>(bufC, bufB, (long long)NC3 * HID);
    seg_sum_edges<<<nblk((long long)EI3 * HID, 256), 256, 0, stream>>>(bufC, I(61), I(62), bufB, EI3, HID);
    launch_gemm(bufB, NC3, HID, F(43), HID, F(44), bufD, HID, 0, 0, stream);
    bn_stats<<<HID, 256, 0, stream>>>(bufD, NC3, stats);
    bn_apply<<<nblk((long long)NC3 * HID, 256), 256, 0, stream>>>(bufD, NC3, stats, F(42), F(41));
    launch_gemm(bufD, NC3, HID, F(45), HID, F(46), hC3, HID, 0, 1, stream);

    // ================= readout + output MLP =================
    colsum<<<512, 256, 0, stream>>>(hA, NA, 512, red);
    colsum<<<256, 256, 0, stream>>>(hC2, NC2, 256, red + 512);
    colsum<<<256, 256, 0, stream>>>(hC3, NC3, 256, red + 768);
    out_mlp<<<1, 256, 0, stream>>>(red, F(47), F(48), F(49), F(50), (float*)d_out);
}
